// VQVAE_816043786377
// MI455X (gfx1250) — matrix operand order repaired and checked by reading, not compiled, not run on hardware
//
#include <hip/hip_runtime.h>

// ---------------------------------------------------------------------------
// VQ-VAE vector quantization for MI455X (gfx1250, wave32, WMMA + TDM).
//
// z: [32,64,64,64] f32 NCHW, codebook: [512,64] f32.
// out: quantized [32,64,64,64] f32 NCHW (8388608 floats) ++ loss scalar.
//
//  - z tile (64 d-lines x 128 rows) DMA'd into LDS by the Tensor Data Mover
//    (tensor_load_to_lds) with 1-DWORD/128-DWORD padding -> 129-float line
//    stride, bank-conflict free. Tracked with TENSORcnt.
//  - S = zf @ cbT via v_wmma_f32_16x16x32_f16 (f16 in, f32 acc), 2 tiles /
//    iteration to hide WMMA->VALU hazards.
//  - argmin_k(||c||^2 - 2S) fused into the tile loop, shfl reduction.
//  - fp32 codebook gather, coalesced NCHW float4 stores, loss = 1.25*MSE.
// ---------------------------------------------------------------------------

typedef _Float16 v16h __attribute__((ext_vector_type(16)));
typedef _Float16 v8h  __attribute__((ext_vector_type(8)));
typedef float    v8f  __attribute__((ext_vector_type(8)));
typedef float    v4f  __attribute__((ext_vector_type(4)));
typedef unsigned int v4u __attribute__((ext_vector_type(4)));
typedef int      v8i  __attribute__((ext_vector_type(8)));
typedef int      v4i  __attribute__((ext_vector_type(4)));

#define BATCH   32
#define DIM     64
#define HW      4096                  // 64*64
#define KCODES  512
#define NTOT    (BATCH * HW)          // 131072 rows
#define ROWS    128                   // rows per workgroup
#define THREADS 256                   // 8 wave32 waves, 16 rows each
#define CB_STRIDE 72                  // halves; 144B row stride -> conflict-free b128
#define Z_LSTRIDE 129                 // floats per d-line (128 + 1 TDM pad DWORD)

// LDS carve-up (dynamic shared memory)
#define CB_OFF    0                                   // 512*72*2       = 73728
#define Z_OFF     73728                               // 64*129*4 (+pad)= 33024
#define CNORM_OFF 106752                              // 512*4          = 2048
#define RIDX_OFF  108800                              // 128*4          = 512
#define RED_OFF   109312                              // 8*4            = 32
#define LDS_BYTES 109344

#define LOSS_SCALE (1.25f / (float)(NTOT * DIM))

extern "C" __global__ void vq_zero_loss(float* __restrict__ loss) {
    *loss = 0.0f;
}

extern "C" __global__ void __launch_bounds__(THREADS)
vq_kernel(const float* __restrict__ z, const float* __restrict__ cb,
          float* __restrict__ out, float* __restrict__ loss)
{
    extern __shared__ char smem[];
    _Float16* cb_lds  = (_Float16*)(smem + CB_OFF);   // [512][72] f16
    float*    z2      = (float*)   (smem + Z_OFF);    // [64 d][129] f32 (TDM-padded)
    float*    cnorm   = (float*)   (smem + CNORM_OFF);// [512]
    int*      row_idx = (int*)     (smem + RIDX_OFF); // [128]
    float*    red     = (float*)   (smem + RED_OFF);  // [8]

    const int t          = threadIdx.x;
    const int block_base = blockIdx.x * ROWS;         // flat NHWC row base
    const int b          = block_base >> 12;          // / HW (blocks 128-aligned)
    const int pos        = block_base & (HW - 1);
    const size_t zb      = (size_t)b * DIM * HW;

    // ---- TDM: DMA z tile (64 lines x 128 floats, NCHW d-major) -> LDS -----
    // D# group0: count=1 | lds_addr | global_addr(57b) | type=2 (bits 127:126)
    // D# group1: data_size=4B, pad_enable, pad_interval=128 DW, pad_amount=1 DW,
    //            tensor_dim0=4096, tensor_dim1=64, tile=128x64, dim0_stride=4096
    if (t < 32) {   // wave 0 only (TDM ignores EXEC; one issue per block)
        unsigned long long ga = (unsigned long long)(const void*)(z + zb + pos);
        v4u g0 = { 1u, (unsigned)Z_OFF,
                   (unsigned)(ga & 0xffffffffull),
                   (unsigned)((ga >> 32) & 0x01ffffffull) | 0x80000000u };
        v8i g1 = { (int)((2u << 16) | (1u << 20) | (6u << 22)),
                   (int)(4096u << 16),          // tensor_dim0[15:0] @ bits 63:48
                   (int)(64u << 16),            // tensor_dim1[15:0] @ bits 95:80
                   (int)(128u << 16),           // tile_dim0 @ bits 127:112
                   64,                          // tile_dim1 @ bits 143:128
                   4096,                        // tensor_dim0_stride low32
                   0, 0 };
        v4i g2 = { 0, 0, 0, 0 };
        v4i g3 = { 0, 0, 0, 0 };
#if defined(__clang_major__) && __clang_major__ >= 23
        v8i g4 = { 0, 0, 0, 0, 0, 0, 0, 0 };
        __builtin_amdgcn_tensor_load_to_lds(g0, g1, g2, g3, g4, 0);
#else
        __builtin_amdgcn_tensor_load_to_lds(g0, g1, g2, g3, 0);
#endif
    }

    // ---- stage codebook -> f16 LDS (coalesced float4 global loads) --------
    {
        const v4f* cb4 = (const v4f*)cb;
        #pragma unroll
        for (int i = 0; i < (KCODES * DIM / 4) / THREADS; ++i) {   // 32 iters
            int e4   = i * THREADS + t;
            v4f v    = cb4[e4];
            int base = e4 << 2;
            int code = base >> 6;
            int d    = base & 63;
            _Float16* p = cb_lds + code * CB_STRIDE + d;
            p[0] = (_Float16)v.x; p[1] = (_Float16)v.y;
            p[2] = (_Float16)v.z; p[3] = (_Float16)v.w;
        }
    }
    if (t < 32) __builtin_amdgcn_s_wait_tensorcnt(0);   // z tile landed in LDS
    __syncthreads();

    // ---- codebook norms from the f16-rounded values (matches WMMA math) ---
    for (int c = t; c < KCODES; c += THREADS) {
        const _Float16* p = cb_lds + c * CB_STRIDE;
        float s = 0.0f;
        #pragma unroll
        for (int j = 0; j < DIM / 8; ++j) {
            v8h v = *(const v8h*)(p + j * 8);
            #pragma unroll
            for (int e = 0; e < 8; ++e) { float f = (float)v[e]; s = fmaf(f, f, s); }
        }
        cnorm[c] = s;
    }
    __syncthreads();

    // ---- per-wave WMMA distance sweep -------------------------------------
    const int wave     = t >> 5;
    const int lane     = t & 31;
    const int r        = lane & 15;       // column (code-within-tile) / A-row
    const int g        = lane >> 4;       // half-wave group
    const int row_base = wave * 16;
    const int myrow    = row_base + r;

    // A fragments (16x32 f16: lane<16 holds K=0..15; lane>=16 K=16..31)
    v16h a0, a1;
    #pragma unroll
    for (int j = 0; j < 8; ++j) {
        a0[j]     = (_Float16)z2[(      16 * g + j) * Z_LSTRIDE + myrow];
        a0[j + 8] = (_Float16)z2[(8 +  16 * g + j) * Z_LSTRIDE + myrow];
        a1[j]     = (_Float16)z2[(32 +  16 * g + j) * Z_LSTRIDE + myrow];
        a1[j + 8] = (_Float16)z2[(40 +  16 * g + j) * Z_LSTRIDE + myrow];
    }

    float bv[8];
    int   bi[8];
    #pragma unroll
    for (int i = 0; i < 8; ++i) { bv[i] = 3.4e38f; bi[i] = 0x7fffffff; }

    for (int tp = 0; tp < KCODES / 32; ++tp) {           // 16 iterations, 2 tiles each
        const int tile0 = tp * 2, tile1 = tp * 2 + 1;
        const _Float16* cbr0 = cb_lds + (tile0 * 16 + r) * CB_STRIDE;
        const _Float16* cbr1 = cb_lds + (tile1 * 16 + r) * CB_STRIDE;
        // B 32x16 f16: lane<16 holds K=0..15 of column n=lane; lane>=16 K=16..31
        v8h p0 = *(const v8h*)(cbr0 + 16 * g);
        v8h p1 = *(const v8h*)(cbr0 + 16 * g + 8);
        v8h p2 = *(const v8h*)(cbr0 + 32 + 16 * g);
        v8h p3 = *(const v8h*)(cbr0 + 32 + 16 * g + 8);
        v8h p4 = *(const v8h*)(cbr1 + 16 * g);
        v8h p5 = *(const v8h*)(cbr1 + 16 * g + 8);
        v8h p6 = *(const v8h*)(cbr1 + 32 + 16 * g);
        v8h p7 = *(const v8h*)(cbr1 + 32 + 16 * g + 8);
        v16h b00 = __builtin_shufflevector(p0, p1, 0,1,2,3,4,5,6,7,8,9,10,11,12,13,14,15);
        v16h b01 = __builtin_shufflevector(p2, p3, 0,1,2,3,4,5,6,7,8,9,10,11,12,13,14,15);
        v16h b10 = __builtin_shufflevector(p4, p5, 0,1,2,3,4,5,6,7,8,9,10,11,12,13,14,15);
        v16h b11 = __builtin_shufflevector(p6, p7, 0,1,2,3,4,5,6,7,8,9,10,11,12,13,14,15);

        v8f c0 = {0.f,0.f,0.f,0.f,0.f,0.f,0.f,0.f};
        v8f c1 = {0.f,0.f,0.f,0.f,0.f,0.f,0.f,0.f};
        c0 = __builtin_amdgcn_wmma_f32_16x16x32_f16(false, a0, false, b00, (short)0, c0, false, false);
        c1 = __builtin_amdgcn_wmma_f32_16x16x32_f16(false, a0, false, b10, (short)0, c1, false, false);
        c0 = __builtin_amdgcn_wmma_f32_16x16x32_f16(false, a1, false, b01, (short)0, c0, false, false);
        c1 = __builtin_amdgcn_wmma_f32_16x16x32_f16(false, a1, false, b11, (short)0, c1, false, false);

        const int   n0  = tile0 * 16 + r;    // C layout: lane = column n
        const int   n1  = tile1 * 16 + r;
        const float cn0 = cnorm[n0];
        const float cn1 = cnorm[n1];
        #pragma unroll
        for (int i = 0; i < 8; ++i) {        // VGPR i = row (8g + i)
            float v0 = fmaf(-2.0f, c0[i], cn0);
            float v1 = fmaf(-2.0f, c1[i], cn1);
            if (v0 < bv[i]) { bv[i] = v0; bi[i] = n0; }
            if (v1 < bv[i]) { bv[i] = v1; bi[i] = n1; }
        }
    }

    // ---- argmin across the 16 columns of each half-wave -------------------
    #pragma unroll
    for (int i = 0; i < 8; ++i) {
        #pragma unroll
        for (int m = 1; m < 16; m <<= 1) {
            float ov   = __shfl_xor(bv[i], m, 32);
            int   oi   = __shfl_xor(bi[i], m, 32);
            bool  take = ov < bv[i];
            bv[i] = take ? ov : bv[i];
            bi[i] = take ? oi : bi[i];
        }
    }
    if (r == 0) {
        #pragma unroll
        for (int i = 0; i < 8; ++i) row_idx[row_base + 8 * g + i] = bi[i];
    }
    __syncthreads();

    // ---- gather fp32 codebook rows, write NCHW, accumulate loss -----------
    float lsum = 0.0f;
    #pragma unroll
    for (int i = 0; i < (ROWS * DIM / 4) / THREADS; ++i) {         // 8 iters
        int e4  = (i * THREADS + t) << 2;
        int d   = e4 >> 7;
        int row = e4 & 127;
        v4f q;
        float* qp = (float*)&q;
        #pragma unroll
        for (int j = 0; j < 4; ++j) {
            int   idxr = row_idx[row + j];
            float qv   = cb[idxr * DIM + d];
            float diff = z2[d * Z_LSTRIDE + row + j] - qv;
            lsum = fmaf(diff, diff, lsum);
            qp[j] = qv;
        }
        *(v4f*)(out + zb + (size_t)d * HW + pos + row) = q;        // coalesced
    }
    #pragma unroll
    for (int m = 1; m < 32; m <<= 1) lsum += __shfl_xor(lsum, m, 32);
    if (lane == 0) red[wave] = lsum;
    __syncthreads();
    if (t == 0) {
        float s = 0.0f;
        #pragma unroll
        for (int w = 0; w < THREADS / 32; ++w) s += red[w];
        atomicAdd(loss, s * LOSS_SCALE);
    }
}

extern "C" void kernel_launch(void* const* d_in, const int* in_sizes, int n_in,
                              void* d_out, int out_size, void* d_ws, size_t ws_size,
                              hipStream_t stream) {
    const float* z  = (const float*)d_in[0];
    const float* cb = (const float*)d_in[1];
    float* out  = (float*)d_out;
    float* loss = out + (size_t)NTOT * DIM;   // quantized first, loss scalar last

    // Allow >64KB dynamic LDS (host-side attribute, not a stream op; capture-safe).
    (void)hipFuncSetAttribute(reinterpret_cast<const void*>(vq_kernel),
                              hipFuncAttributeMaxDynamicSharedMemorySize, LDS_BYTES);

    vq_zero_loss<<<dim3(1), dim3(1), 0, stream>>>(loss);
    vq_kernel<<<dim3(NTOT / ROWS), dim3(THREADS), LDS_BYTES, stream>>>(z, cb, out, loss);
}
